// AttentionBlock_76201309765932
// MI455X (gfx1250) — matrix operand order repaired and checked by reading, not compiled, not run on hardware
//
#include <hip/hip_runtime.h>
#include <hip/hip_bf16.h>
#include <math.h>

typedef __attribute__((ext_vector_type(16))) _Float16 v16h;
typedef __attribute__((ext_vector_type(8)))  _Float16 v8h;
typedef __attribute__((ext_vector_type(8)))  float    v8f;
typedef __attribute__((ext_vector_type(4)))  unsigned int u32x4;

#define C_DIM 512
#define T_DIM 2304
#define B_DIM 4
#define NH    8
#define CH    64
#define NG    32
#define GSZ   (C_DIM / NG)        // 16
#define GN_N  (GSZ * T_DIM)       // 36864
#define QK_SCALE 0.35355339059327373f   // 1/sqrt(sqrt(64))
#define GN_EPS 1e-5f

static __device__ __forceinline__ int lane32() { return (int)(threadIdx.x & 31u); }

static __device__ __forceinline__ v8f wmma_f16(v16h a, v16h b, v8f c) {
  // D = A(16x32,f16) * B(32x16,f16) + C(f32)
  return __builtin_amdgcn_wmma_f32_16x16x32_f16(false, a, false, b, (short)0, c,
                                                false, false);
}

// ---- CDNA5 async global->LDS copy (ASYNCcnt-tracked), 16B per lane ---------
static __device__ __forceinline__ unsigned lds_off_of(const void* p) {
  // generic LDS pointer: low 32 bits = LDS byte offset (aperture truncation)
  return (unsigned)(unsigned long long)(uintptr_t)p;
}
static __device__ __forceinline__ void async_copy_b128(unsigned lds_byte_off,
                                                       const void* gaddr) {
  asm volatile("global_load_async_to_lds_b128 %0, %1, off"
               :: "v"(lds_byte_off), "v"((unsigned long long)(uintptr_t)gaddr)
               : "memory");
}
static __device__ __forceinline__ void wait_async_le2() {
  asm volatile("s_wait_asynccnt 0x2" ::: "memory");
}
static __device__ __forceinline__ void wait_async_0() {
  asm volatile("s_wait_asynccnt 0x0" ::: "memory");
}

// A fragment (16 rows x K=32, f16) from row-major f16 src with leading dim ld.
// Contiguous k order (same as load_b_T): lane r=l&15 row M=r; lanes<16 hold K=0..15, lanes>=16 K=16..31.
static __device__ __forceinline__ v16h load_a_f16(const _Float16* src, int ld) {
  int l = lane32();
  const _Float16* row = src + (size_t)(l & 15) * ld + ((l >> 4) << 4);
  union { v16h v; v8h h[2]; } u;
  u.h[0] = *(const v8h*)(row);
  u.h[1] = *(const v8h*)(row + 8);
  return u.v;
}

// B fragment (K=32 x 16 cols, f16) from "transposed" source srcT[N][K] (K contiguous).
// Contiguous k order (same as load_a_f16): lane's column N=l&15; lanes<16 hold K=0..15, lanes>=16 hold K=16..31.
static __device__ __forceinline__ v16h load_b_T(const _Float16* srcT, int ld) {
  int l = lane32();
  return *(const v16h*)(srcT + (size_t)(l & 15) * ld + ((l >> 4) << 4));
}

// ---------------------------------------------------------------------------
// Kernel 0: one-time f32 -> f16 weight conversion (qkv_w, proj_w)
// ---------------------------------------------------------------------------
__global__ void __launch_bounds__(256) cvtw_kernel(const float* __restrict__ w,
                                                   _Float16* __restrict__ o,
                                                   int n4) {
  int i = blockIdx.x * 256 + (int)threadIdx.x;
  if (i < n4) {
    float4 v = ((const float4*)w)[i];
    union { _Float16 h[4]; unsigned long long u; } pk;
    pk.h[0] = (_Float16)v.x; pk.h[1] = (_Float16)v.y;
    pk.h[2] = (_Float16)v.z; pk.h[3] = (_Float16)v.w;
    *(unsigned long long*)(o + (size_t)i * 4) = pk.u;
  }
}

// ---------------------------------------------------------------------------
// Kernel 1: GroupNorm(32 groups) -> xn (f32, [b][c][t]) and xnT (f16, [b][t][c])
// ---------------------------------------------------------------------------
__global__ void __launch_bounds__(256) gn_kernel(
    const float* __restrict__ x, const float* __restrict__ gw,
    const float* __restrict__ gb, float* __restrict__ xn,
    _Float16* __restrict__ xnT) {
  int b = blockIdx.x / NG, g = blockIdx.x % NG;
  const float* xp = x + ((size_t)b * C_DIM + (size_t)g * GSZ) * T_DIM;
  float s = 0.f, ss = 0.f;
  for (int i = threadIdx.x; i < GN_N; i += 256) {
    float v = xp[i];
    s += v; ss += v * v;
  }
  __shared__ float r1[256], r2[256];
  r1[threadIdx.x] = s; r2[threadIdx.x] = ss;
  __syncthreads();
  for (int off = 128; off > 0; off >>= 1) {
    if ((int)threadIdx.x < off) {
      r1[threadIdx.x] += r1[threadIdx.x + off];
      r2[threadIdx.x] += r2[threadIdx.x + off];
    }
    __syncthreads();
  }
  float mean = r1[0] * (1.0f / GN_N);
  float var  = r2[0] * (1.0f / GN_N) - mean * mean;
  float rstd = rsqrtf(var + GN_EPS);
  float* xnp = xn + ((size_t)b * C_DIM + (size_t)g * GSZ) * T_DIM;
  _Float16* xTp = xnT + (size_t)b * T_DIM * C_DIM + (size_t)g * GSZ;
  for (int i = threadIdx.x; i < GN_N; i += 256) {
    int cl = i / T_DIM, t = i % T_DIM;
    int c = g * GSZ + cl;
    float y = (xp[i] - mean) * rstd * gw[c] + gb[c];
    xnp[i] = y;
    xTp[(size_t)t * C_DIM + cl] = (_Float16)y;
  }
}

// ---------------------------------------------------------------------------
// Kernel 2: QKV 1x1 conv (GEMM 1536x512 @ 512xT per batch), one wave per
// 16x16 tile. Writes qT/kT ([bh][t][64], pre-scaled) and v ([bh][64][T]).
// ---------------------------------------------------------------------------
__global__ void __launch_bounds__(256) qkv_kernel(
    const _Float16* __restrict__ W16, const float* __restrict__ bias,
    const _Float16* __restrict__ xnT,
    _Float16* __restrict__ qT, _Float16* __restrict__ kT,
    _Float16* __restrict__ vW) {
  int tile = blockIdx.x * 8 + (int)(threadIdx.x >> 5);
  int nt = tile % 144; int r = tile / 144; int mt = r % 96; int b = r / 96;
  int m0 = mt * 16, n0 = nt * 16;
  const _Float16* Arow = W16 + (size_t)m0 * C_DIM;
  const _Float16* Bt = xnT + ((size_t)b * T_DIM + n0) * C_DIM;

  v8f acc = {};
#pragma unroll 4
  for (int kk = 0; kk < C_DIM; kk += 32) {
    __builtin_prefetch(Arow + kk + 256, 0, 1);
    acc = wmma_f16(load_a_f16(Arow + kk, C_DIM), load_b_T(Bt + kk, C_DIM), acc);
  }

  int l = lane32(); int n = l & 15; int half = l >> 4;
  const float* bp = bias + m0 + half * 8;   // bias for rows M = 8*half + v
  int hh = m0 / 192;                        // head within batch
  int sub = m0 % 192;                       // 0..63 q, 64..127 k, 128..191 v
  int bh = b * NH + hh;

  if (sub < 128) {
    _Float16* base = (sub < 64) ? qT : kT;
    int c0 = (sub < 64) ? sub : sub - 64;
    _Float16* dst = base + ((size_t)bh * T_DIM + n0 + n) * CH + c0 + half * 8;
    union { _Float16 h[8]; u32x4 u; } pk;
#pragma unroll
    for (int v = 0; v < 8; ++v) pk.h[v] = (_Float16)((acc[v] + bp[v]) * QK_SCALE);
    *(u32x4*)dst = pk.u;                    // 16B contiguous transposed store
  } else {
    int c0 = sub - 128;
    _Float16* dst = vW + ((size_t)bh * CH + c0 + half * 8) * T_DIM + n0 + n;
#pragma unroll
    for (int v = 0; v < 8; ++v) dst[(size_t)v * T_DIM] = (_Float16)(acc[v] + bp[v]);
  }
}

// ---------------------------------------------------------------------------
// Kernel 3: flash attention. Block = 8 waves; each wave owns a 16-row t-tile.
// K/V tiles are shared by all 8 waves -> double-buffered async global->LDS
// staging (ASYNCcnt). Online softmax, s streamed in chunks of 32.
// Output aT: [b][t][512] f16.
// ---------------------------------------------------------------------------
struct Slab {
  float    S[16][36];     // padded f32 score tile (16 x 32)
  _Float16 P[16][32];     // exp() probabilities, in A-fragment row-major form
  float    frow[16];      // per-row rescale factor
  float    lrow[16];      // per-row running denominator
};

__global__ void __launch_bounds__(256) attn_kernel(
    const _Float16* __restrict__ qT, const _Float16* __restrict__ kT,
    const _Float16* __restrict__ vW, _Float16* __restrict__ aT) {
  __shared__ Slab slabs[8];
  __shared__ __align__(32) _Float16 kTile[2][32][CH];   // [s][c] slice, 4KB each
  __shared__ __align__(32) _Float16 vTile[2][CH][32];   // [c][s] slice, 4KB each

  int w = (int)(threadIdx.x >> 5);
  Slab& sl = slabs[w];
  int bh = blockIdx.y;
  int t0 = blockIdx.x * 128 + w * 16;
  int b = bh / NH, hh = bh % NH;
  int l = lane32(); int n = l & 15; int half = l >> 4;
  int tid = (int)threadIdx.x;

  const _Float16* kbh = kT + (size_t)bh * T_DIM * CH;       // [s][64] rows
  const _Float16* vbh = vW + (size_t)bh * CH * T_DIM;       // [c][T] rows

  // stage K(32x64) + V(64x32) slice for s-chunk into LDS buffer `buf`
  auto stage = [&](int buf, int s) {
    // K slice: fully contiguous 4KB; 256 threads x 16B
    async_copy_b128(lds_off_of(&kTile[buf][0][0]) + tid * 16,
                    (const char*)(kbh + (size_t)s * CH) + tid * 16);
    // V slice: 64 rows x 64B (row stride T*2 bytes); 4 threads per row
    async_copy_b128(lds_off_of(&vTile[buf][0][0]) + (tid >> 2) * 64 + (tid & 3) * 16,
                    (const char*)(vbh + (size_t)(tid >> 2) * T_DIM + s) + (tid & 3) * 16);
  };

  const _Float16* qbase = qT + ((size_t)bh * T_DIM + t0) * CH;
  v16h qf0 = load_a_f16(qbase, CH);        // K = c 0..31
  v16h qf1 = load_a_f16(qbase + 32, CH);   // K = c 32..63

  v8f acc0 = {}, acc1 = {}, acc2 = {}, acc3 = {};
  float mrun = -INFINITY, lrun = 0.f;

  stage(0, 0);                              // pre-issue first chunk (cnt=2)

  for (int s0 = 0, it = 0; s0 < T_DIM; s0 += 32, ++it) {
    int buf = it & 1;
    bool has_next = (s0 + 32) < T_DIM;
    if (has_next) { stage(buf ^ 1, s0 + 32); wait_async_le2(); }
    else          { wait_async_0(); }
    __syncthreads();                        // staged K/V visible to all waves

    // ---- S = (q*sc)^T (k*sc) for two 16-wide s sub-tiles --------------------
    const _Float16* kl = &kTile[buf][0][0];
    v8f S0 = {}, S1 = {};
    S0 = wmma_f16(qf0, load_b_T(kl, CH), S0);
    S0 = wmma_f16(qf1, load_b_T(kl + 32, CH), S0);
    S1 = wmma_f16(qf0, load_b_T(kl + 16 * CH, CH), S1);
    S1 = wmma_f16(qf1, load_b_T(kl + 16 * CH + 32, CH), S1);
#pragma unroll
    for (int v = 0; v < 8; ++v) {
      sl.S[half * 8 + v][n]      = S0[v];
      sl.S[half * 8 + v][16 + n] = S1[v];
    }
    __syncthreads();

    // ---- online softmax: lanes 0..15 each own one t-row ---------------------
    if (l < 16) {
      float row[32]; float mx = -INFINITY;
#pragma unroll
      for (int j = 0; j < 32; ++j) { row[j] = sl.S[l][j]; mx = fmaxf(mx, row[j]); }
      float mnew = fmaxf(mrun, mx);
      float f = __expf(mrun - mnew);
      float sum = 0.f;
#pragma unroll
      for (int j = 0; j < 32; ++j) {
        float p = __expf(row[j] - mnew);
        sum += p;
        sl.P[l][j] = (_Float16)p;
      }
      lrun = lrun * f + sum;
      mrun = mnew;
      sl.frow[l] = f;
    }
    __syncthreads();

    // ---- rescale accumulators, then acc += P @ V^T --------------------------
    float fm[8];
#pragma unroll
    for (int v = 0; v < 8; ++v) fm[v] = sl.frow[half * 8 + v];
#pragma unroll
    for (int v = 0; v < 8; ++v) {
      acc0[v] *= fm[v]; acc1[v] *= fm[v]; acc2[v] *= fm[v]; acc3[v] *= fm[v];
    }
    v16h pf = load_a_f16(&sl.P[0][0], 32);
    const _Float16* vl = &vTile[buf][0][0];
    acc0 = wmma_f16(pf, load_b_T(vl, 32), acc0);
    acc1 = wmma_f16(pf, load_b_T(vl + 16 * 32, 32), acc1);
    acc2 = wmma_f16(pf, load_b_T(vl + 32 * 32, 32), acc2);
    acc3 = wmma_f16(pf, load_b_T(vl + 48 * 32, 32), acc3);
    __syncthreads();                        // all reads done before next refill
  }

  if (l < 16) sl.lrow[l] = lrun;
  __syncthreads();
  float li[8];
#pragma unroll
  for (int v = 0; v < 8; ++v) li[v] = 1.f / sl.lrow[half * 8 + v];

  _Float16* out = aT + ((size_t)b * T_DIM + t0) * C_DIM + (size_t)hh * CH;
#pragma unroll
  for (int v = 0; v < 8; ++v) {
    _Float16* orow = out + (size_t)(half * 8 + v) * C_DIM;
    orow[n]      = (_Float16)(acc0[v] * li[v]);
    orow[16 + n] = (_Float16)(acc1[v] * li[v]);
    orow[32 + n] = (_Float16)(acc2[v] * li[v]);
    orow[48 + n] = (_Float16)(acc3[v] * li[v]);
  }
}

// ---------------------------------------------------------------------------
// Kernel 4: proj GEMM (512x512 @ 512xT) + bias + residual (normalized x)
// ---------------------------------------------------------------------------
__global__ void __launch_bounds__(256) proj_kernel(
    const _Float16* __restrict__ W16, const float* __restrict__ bias,
    const _Float16* __restrict__ aT, const float* __restrict__ xn,
    float* __restrict__ out) {
  int tile = blockIdx.x * 8 + (int)(threadIdx.x >> 5);
  int nt = tile % 144; int r = tile / 144; int mt = r % 32; int b = r / 32;
  int m0 = mt * 16, n0 = nt * 16;
  const _Float16* Arow = W16 + (size_t)m0 * C_DIM;
  const _Float16* Bt = aT + ((size_t)b * T_DIM + n0) * C_DIM;

  v8f acc = {};
#pragma unroll 4
  for (int kk = 0; kk < C_DIM; kk += 32) {
    __builtin_prefetch(Arow + kk + 256, 0, 1);
    acc = wmma_f16(load_a_f16(Arow + kk, C_DIM), load_b_T(Bt + kk, C_DIM), acc);
  }

  int l = lane32(); int n = l & 15; int half = l >> 4;
  const float* bp  = bias + m0 + half * 8;
  const float* xnp = xn  + ((size_t)b * C_DIM + m0 + half * 8) * T_DIM + n0 + n;
  float*       op  = out + ((size_t)b * C_DIM + m0 + half * 8) * T_DIM + n0 + n;
#pragma unroll
  for (int v = 0; v < 8; ++v) {
    op[(size_t)v * T_DIM] = acc[v] + bp[v] + xnp[(size_t)v * T_DIM];
  }
}

// ---------------------------------------------------------------------------
extern "C" void kernel_launch(void* const* d_in, const int* in_sizes, int n_in,
                              void* d_out, int out_size, void* d_ws, size_t ws_size,
                              hipStream_t stream) {
  const float* x    = (const float*)d_in[0];
  const float* gw   = (const float*)d_in[1];
  const float* gb   = (const float*)d_in[2];
  const float* qkvw = (const float*)d_in[3];
  const float* qkvb = (const float*)d_in[4];
  const float* pw   = (const float*)d_in[5];
  const float* pb   = (const float*)d_in[6];
  float* out = (float*)d_out;

  const size_t NE = (size_t)B_DIM * C_DIM * T_DIM;   // 4,718,592 elements
  float*    xn   = (float*)d_ws;                     // [b][c][t] f32 (residual)
  _Float16* xnT  = (_Float16*)(xn + NE);             // [b][t][c] f16
  _Float16* qT   = xnT + NE;                         // [bh][t][64] f16 (scaled)
  _Float16* kT   = qT + NE;                          // [bh][s][64] f16 (scaled)
  _Float16* vW   = kT + NE;                          // [bh][c][s] f16
  _Float16* aT   = vW + NE;                          // [b][t][512] f16
  _Float16* wq16 = aT + NE;                          // 1536x512 f16
  _Float16* wp16 = wq16 + (size_t)3 * C_DIM * C_DIM; // 512x512 f16

  cvtw_kernel<<<dim3(768), dim3(256), 0, stream>>>(qkvw, wq16, 3 * C_DIM * C_DIM / 4);
  cvtw_kernel<<<dim3(256), dim3(256), 0, stream>>>(pw, wp16, C_DIM * C_DIM / 4);
  gn_kernel  <<<dim3(B_DIM * NG), dim3(256), 0, stream>>>(x, gw, gb, xn, xnT);
  qkv_kernel <<<dim3(6912),       dim3(256), 0, stream>>>(wq16, qkvb, xnT, qT, kT, vW);
  attn_kernel<<<dim3(18, 32),     dim3(256), 0, stream>>>(qT, kT, vW, aT);
  proj_kernel<<<dim3(2304),       dim3(256), 0, stream>>>(wp16, pb, aT, xn, out);
}
